// Graph_STS_10093173145811
// MI455X (gfx1250) — compile-verified, matrix operand order checked by reading, not run on hardware
//
#include <hip/hip_runtime.h>
#include <hip/hip_fp16.h>

typedef __attribute__((ext_vector_type(16))) _Float16 v16h;
typedef __attribute__((ext_vector_type(8)))  float    v8f;

static constexpr int Bsz = 256;
static constexpr int NX  = 64;
static constexpr int NY  = 64;
static constexpr int NP  = 32;
static constexpr int D   = 256;

// ---------------------------------------------------------------------------
// Convert W1/W2 (f32, [D,D] row-major, k-major rows) into WMMA B-fragment
// order (f16): frag (kt,nt) -> 512 halves, lane L holds B[kt*32+(L/16)*16+j,
// nt*16+(L%16)] for j=0..15 contiguous.  One 32B load per lane per fragment.
// ---------------------------------------------------------------------------
__global__ void prep_weights(const float* __restrict__ w1,
                             const float* __restrict__ w2,
                             _Float16* __restrict__ w1h,
                             _Float16* __restrict__ w2h) {
  int idx = blockIdx.x * blockDim.x + threadIdx.x;   // 0 .. 2*65536-1
  const float* W   = (idx & 65536) ? w2 : w1;
  _Float16*    Wh  = (idx & 65536) ? w2h : w1h;
  int e      = idx & 65535;
  int frag   = e >> 9;          // 0..127  (kt*16 + nt)
  int within = e & 511;
  int lane   = within >> 4;     // 0..31
  int j      = within & 15;     // 0..15
  int kt = frag >> 4;           // 0..7
  int nt = frag & 15;           // 0..15
  int k = kt * 32 + ((lane >> 4) << 4) + j;
  int n = (nt << 4) + (lane & 15);
  Wh[e] = (_Float16)W[k * D + n];
}

// ---------------------------------------------------------------------------
// Per-sample gather + max-pool over NX / NY node rows.  One block per sample,
// one thread per feature column (fully coalesced 1KB row loads).
// ---------------------------------------------------------------------------
__global__ void gather_maxpool(const float* __restrict__ feat,
                               const int* __restrict__ x_id,
                               const int* __restrict__ y_id,
                               float* __restrict__ arga,
                               float* __restrict__ argb,
                               float* __restrict__ argc_out) {
  int b = blockIdx.x;
  int t = threadIdx.x;

  float ma = -__builtin_inff();
  for (int i = 0; i < NX; ++i) {
    size_t id = (size_t)(unsigned)x_id[b * NX + i];
    if (i + 1 < NX) {
      size_t nid = (size_t)(unsigned)x_id[b * NX + i + 1];
      __builtin_prefetch(feat + nid * D + t, 0, 0);   // global_prefetch_b8
    }
    ma = fmaxf(ma, feat[id * D + t]);
  }
  arga[b * D + t] = ma;

  float mb = -__builtin_inff();
  for (int i = 0; i < NY; ++i) {
    size_t id = (size_t)(unsigned)y_id[b * NY + i];
    if (i + 1 < NY) {
      size_t nid = (size_t)(unsigned)y_id[b * NY + i + 1];
      __builtin_prefetch(feat + nid * D + t, 0, 0);
    }
    mb = fmaxf(mb, feat[id * D + t]);
  }
  argb[b * D + t] = mb;
  argc_out[b * D + t] = mb;    // argc = argb.reshape(-1)
}

// ---------------------------------------------------------------------------
// Attention reg-loss: 64 dot products (32 L + 32 R) per sample, softmax over
// NP, mean of squared weights.  Wave32 shuffle butterflies; per-sample partial
// written to workspace (deterministic tree reduce later, no float atomics).
// ---------------------------------------------------------------------------
__global__ void attention_reg(const float* __restrict__ feat,
                              const int* __restrict__ plo,
                              const int* __restrict__ pro,
                              const float* __restrict__ w6w, const float* __restrict__ w6b,
                              const float* __restrict__ w7w, const float* __restrict__ w7b,
                              float* __restrict__ regpart) {
  __shared__ float sc[64];
  int b    = blockIdx.x;
  int lane = threadIdx.x & 31;
  int wave = threadIdx.x >> 5;

  for (int r = wave; r < 64; r += 8) {
    int side = r >> 5;
    int n    = r & 31;
    size_t id = (size_t)(unsigned)(side ? pro[b * NP + n] : plo[b * NP + n]);
    const float* wv = side ? w7w : w6w;
    float s = 0.f;
#pragma unroll
    for (int j = 0; j < 8; ++j)
      s += feat[id * D + lane + 32 * j] * wv[lane + 32 * j];
    for (int o = 16; o; o >>= 1) s += __shfl_xor(s, o, 32);
    if (lane == 0) sc[r] = s + (side ? w7b[0] : w6b[0]);
  }
  __syncthreads();

  if (threadIdx.x < 32) {
    float acc = 0.f;
#pragma unroll
    for (int side = 0; side < 2; ++side) {
      float v = sc[side * 32 + lane];
      float m = v;
      for (int o = 16; o; o >>= 1) m = fmaxf(m, __shfl_xor(m, o, 32));
      float e = expf(v - m);
      float s = e;
      for (int o = 16; o; o >>= 1) s += __shfl_xor(s, o, 32);
      float at = e / s;
      float sq = at * at;
      for (int o = 16; o; o >>= 1) sq += __shfl_xor(sq, o, 32);
      acc += sq;                       // lane0 holds full sum after butterfly
    }
    if (lane == 0) regpart[b] = 0.5f * acc * (1.0f / (float)NP);
  }
}

// ---------------------------------------------------------------------------
// h = tanh(|a-b|@W1 + (a*b)@W2 + b1 + b2) via v_wmma_f32_16x16x32_f16.
// One wave per 16x16 output tile; 8 K-steps, 2 WMMAs per step.
// ---------------------------------------------------------------------------
__global__ void __launch_bounds__(32)
mlp_wmma(const float* __restrict__ arga,
         const float* __restrict__ argb,
         const _Float16* __restrict__ w1h,
         const _Float16* __restrict__ w2h,
         const float* __restrict__ w1b,
         const float* __restrict__ w2b,
         float* __restrict__ h) {
  int lane = threadIdx.x;              // 0..31, EXEC all ones
  int m0 = blockIdx.x * 16;            // row tile
  int nt = blockIdx.y;                 // col tile 0..15
  int row   = m0 + (lane & 15);
  int khalf = (lane >> 4) << 4;        // lanes 16-31 take K+16

  v8f acc = {};
  for (int kt = 0; kt < 8; ++kt) {
    int kbase = kt * 32 + khalf;
    const float4* pa = (const float4*)(arga + row * D + kbase);
    const float4* pb = (const float4*)(argb + row * D + kbase);
    v16h ax, ay;
#pragma unroll
    for (int q = 0; q < 4; ++q) {
      float4 a4 = pa[q];
      float4 b4 = pb[q];
      ax[q * 4 + 0] = (_Float16)fabsf(a4.x - b4.x);
      ax[q * 4 + 1] = (_Float16)fabsf(a4.y - b4.y);
      ax[q * 4 + 2] = (_Float16)fabsf(a4.z - b4.z);
      ax[q * 4 + 3] = (_Float16)fabsf(a4.w - b4.w);
      ay[q * 4 + 0] = (_Float16)(a4.x * b4.x);
      ay[q * 4 + 1] = (_Float16)(a4.y * b4.y);
      ay[q * 4 + 2] = (_Float16)(a4.z * b4.z);
      ay[q * 4 + 3] = (_Float16)(a4.w * b4.w);
    }
    const v16h* f1 = (const v16h*)(w1h + ((size_t)(kt * 16 + nt) << 9)) + lane;
    const v16h* f2 = (const v16h*)(w2h + ((size_t)(kt * 16 + nt) << 9)) + lane;
    v16h bw1 = *f1;
    v16h bw2 = *f2;
    acc = __builtin_amdgcn_wmma_f32_16x16x32_f16(false, ax, false, bw1,
                                                 (short)0, acc, false, false);
    acc = __builtin_amdgcn_wmma_f32_16x16x32_f16(false, ay, false, bw2,
                                                 (short)0, acc, false, false);
  }

  int n = nt * 16 + (lane & 15);
  float bias = w1b[n] + w2b[n];
  int rbase = m0 + ((lane >> 4) << 3);   // lanes 16-31 hold rows M=8..15
#pragma unroll
  for (int v = 0; v < 8; ++v)
    h[(size_t)(rbase + v) * D + n] = tanhf(acc[v] + bias);
}

// ---------------------------------------------------------------------------
// logits[b, c] = sum_d h[b,d] * cls_w[d,c] + cls_b[c]  (fixed-order LDS tree)
// ---------------------------------------------------------------------------
__global__ void logits_kernel(const float* __restrict__ h,
                              const float* __restrict__ clsw,
                              const float* __restrict__ clsb,
                              float* __restrict__ logits) {
  __shared__ float red[256];
  int b = blockIdx.x;
  int t = threadIdx.x;
  float hv = h[b * D + t];
  for (int c = 0; c < 3; ++c) {
    red[t] = hv * clsw[t * 3 + c];
    __syncthreads();
    for (int o = 128; o; o >>= 1) {
      if (t < o) red[t] += red[t + o];
      __syncthreads();
    }
    if (t == 0) logits[b * 3 + c] = red[0] + clsb[c];
    __syncthreads();
  }
}

// ---------------------------------------------------------------------------
// reg_loss = mean(regpart)  (deterministic tree)
// ---------------------------------------------------------------------------
__global__ void reg_reduce(const float* __restrict__ regpart,
                           float* __restrict__ reg) {
  __shared__ float red[256];
  int t = threadIdx.x;
  red[t] = regpart[t];
  __syncthreads();
  for (int o = 128; o; o >>= 1) {
    if (t < o) red[t] += red[t + o];
    __syncthreads();
  }
  if (t == 0) reg[0] = red[0] * (1.0f / (float)Bsz);
}

extern "C" void kernel_launch(void* const* d_in, const int* in_sizes, int n_in,
                              void* d_out, int out_size, void* d_ws, size_t ws_size,
                              hipStream_t stream) {
  const float* feat  = (const float*)d_in[0];
  const int* x_id    = (const int*)d_in[1];
  const int* y_id    = (const int*)d_in[2];
  const int* plo_id  = (const int*)d_in[3];
  const int* pro_id  = (const int*)d_in[4];
  const float* w1_w  = (const float*)d_in[5];
  const float* w1_b  = (const float*)d_in[6];
  const float* w2_w  = (const float*)d_in[7];
  const float* w2_b  = (const float*)d_in[8];
  const float* w6_w  = (const float*)d_in[9];
  const float* w6_b  = (const float*)d_in[10];
  const float* w7_w  = (const float*)d_in[11];
  const float* w7_b  = (const float*)d_in[12];
  const float* cls_w = (const float*)d_in[13];
  const float* cls_b = (const float*)d_in[14];

  float* out    = (float*)d_out;
  float* logits = out;                  // [256,3]
  float* argc   = out + Bsz * 3;        // [256*256]
  float* reg    = out + Bsz * 3 + Bsz * D;

  float* ws       = (float*)d_ws;
  float* arga     = ws;                       // 65536
  float* argb     = ws + 65536;               // 65536
  float* hbuf     = ws + 131072;              // 65536
  float* regpart  = ws + 196608;              // 256
  _Float16* w1h   = (_Float16*)(ws + 196864); // 65536 halves
  _Float16* w2h   = w1h + 65536;              // 65536 halves

  prep_weights<<<512, 256, 0, stream>>>(w1_w, w2_w, w1h, w2h);
  gather_maxpool<<<Bsz, 256, 0, stream>>>(feat, x_id, y_id, arga, argb, argc);
  attention_reg<<<Bsz, 256, 0, stream>>>(feat, plo_id, pro_id,
                                         w6_w, w6_b, w7_w, w7_b, regpart);
  mlp_wmma<<<dim3(Bsz / 16, D / 16), 32, 0, stream>>>(arga, argb, w1h, w2h,
                                                      w1_b, w2_b, hbuf);
  logits_kernel<<<Bsz, 256, 0, stream>>>(hbuf, cls_w, cls_b, logits);
  reg_reduce<<<1, 256, 0, stream>>>(regpart, reg);
}
